// Mamba_Block_11304353923674
// MI455X (gfx1250) — matrix operand order repaired and checked by reading, not compiled, not run on hardware
//
#include <hip/hip_runtime.h>
#include <hip/hip_bf16.h>
#include <stdint.h>

// ---------------------------------------------------------------------------
// Shapes (fixed by the reference)
// ---------------------------------------------------------------------------
#define D_MODEL 256
#define D_INNER 512
#define D_STATE 16
#define D_CONV  4
#define DT_RANK 16
#define OUT_DIM 1024
#define B_SZ    2
#define SEQ     2048
#define TOKENS  (B_SZ * SEQ)          // 4096

// ---------------------------------------------------------------------------
// WMMA vector types (CDNA5 gfx1250, wave32)
// ---------------------------------------------------------------------------
typedef __attribute__((ext_vector_type(16))) __bf16 v16bf;
typedef __attribute__((ext_vector_type(8)))  float  v8f;

union BFVec {
    v16bf v;
    uint4 q[2];
};

__device__ __forceinline__ uint16_t f2bf(float f) {
    uint32_t u = __float_as_uint(f);
    u += 0x7FFFu + ((u >> 16) & 1u);   // round-to-nearest-even
    return (uint16_t)(u >> 16);
}

__device__ __forceinline__ float block_sum_256(float v, float* red) {
    #pragma unroll
    for (int off = 16; off; off >>= 1) v += __shfl_xor(v, off, 32);
    if ((threadIdx.x & 31) == 0) red[threadIdx.x >> 5] = v;
    __syncthreads();
    float t = 0.f;
    #pragma unroll
    for (int i = 0; i < 8; ++i) t += red[i];
    __syncthreads();
    return t;
}

// ---------------------------------------------------------------------------
// bf16 WMMA GEMM, N-register-blocked + double-buffered K pipeline:
//   C(M,N) f32 = A(M,K) bf16  x  W(N,K) bf16 (row major)
// Each wave computes a 16 x (16*NT) strip (one A tile reused by NT B tiles,
// NT accumulators).  Two named operand sets are software-pipelined:
//     load(k+1) -> WMMA(k) -> load(k+2) -> WMMA(k+1)
// so next-step global_load_b128 clauses are in flight across the WMMAs and
// s_wait_loadcnt can retire partially instead of draining to zero.
// K/32 must be even (true for K = 256 / 512 / 1024 here).
// ---------------------------------------------------------------------------
template<int NT>
__global__ __launch_bounds__(256)
void wmma_gemm_bf16(const uint16_t* __restrict__ A,
                    const uint16_t* __restrict__ W,
                    float* __restrict__ C,
                    int M, int N, int K) {
    int wave    = (blockIdx.x * blockDim.x + threadIdx.x) >> 5;
    int lane    = threadIdx.x & 31;
    int groupsN = (N >> 4) / NT;
    int tileM   = (wave / groupsN) << 4;
    int tileN0  = (wave % groupsN) * (16 * NT);
    if (tileM >= M) return;

    int laneLo = lane & 15;
    int laneHi = lane >> 4;

    v8f acc[NT];
    #pragma unroll
    for (int t = 0; t < NT; ++t) acc[t] = (v8f){};

    const uint16_t* arow = A + (size_t)(tileM + laneLo) * K;
    const uint16_t* wrow[NT];
    #pragma unroll
    for (int t = 0; t < NT; ++t)
        wrow[t] = W + (size_t)(tileN0 + t * 16 + laneLo) * K;

    // A 16x32: lane(m=laneLo) halves 0-7  -> K = k0+laneHi*8 .. +7
    //                          halves 8-15 -> K = k0+16+laneHi*8 .. +7
    // B 32x16: lane(n=laneLo) same K mapping as A
    auto loadAB = [&](BFVec& a, BFVec (&b)[NT], int k0) {
        a.q[0] = *(const uint4*)(arow + k0 + laneHi * 8);
        a.q[1] = *(const uint4*)(arow + k0 + 16 + laneHi * 8);
        #pragma unroll
        for (int t = 0; t < NT; ++t) {
            b[t].q[0] = *(const uint4*)(wrow[t] + k0 + laneHi * 8);
            b[t].q[1] = *(const uint4*)(wrow[t] + k0 + 16 + laneHi * 8);
        }
    };
    auto mmaAB = [&](BFVec& a, BFVec (&b)[NT]) {
        #pragma unroll
        for (int t = 0; t < NT; ++t)
            acc[t] = __builtin_amdgcn_wmma_f32_16x16x32_bf16(
                        false, a.v, false, b[t].v, (short)0, acc[t], false, false);
    };

    int ksteps = K >> 5;              // even for all GEMMs in this model
    BFVec a0, a1;
    BFVec b0[NT], b1[NT];
    loadAB(a0, b0, 0);
    for (int ks = 0; ks < ksteps; ks += 2) {
        loadAB(a1, b1, (ks + 1) << 5);        // prefetch odd step
        mmaAB(a0, b0);                        // consume even step
        if (ks + 2 < ksteps)
            loadAB(a0, b0, (ks + 2) << 5);    // prefetch next even step
        mmaAB(a1, b1);                        // consume odd step
    }

    // D 16x16 f32: element (m = r + laneHi*8, n = laneLo)
    #pragma unroll
    for (int t = 0; t < NT; ++t) {
        float* crow = C + (size_t)tileM * N + tileN0 + t * 16 + laneLo;
        #pragma unroll
        for (int r = 0; r < 8; ++r)
            crow[(size_t)(r + laneHi * 8) * N] = acc[t][r];
    }
}

// ---------------------------------------------------------------------------
// f32 -> bf16 weight conversion
// ---------------------------------------------------------------------------
__global__ __launch_bounds__(256)
void cvt_bf16_kernel(const float* __restrict__ in, uint16_t* __restrict__ out, int n) {
    int i = blockIdx.x * 256 + threadIdx.x;
    if (i < n) out[i] = f2bf(in[i]);
}

// ---------------------------------------------------------------------------
// Embed transpose (B,C,L)->(T,C) + LayerNorm1.  One token per block.
// ---------------------------------------------------------------------------
__global__ __launch_bounds__(256)
void ln_embed_kernel(const float* __restrict__ x,
                     const float* __restrict__ g, const float* __restrict__ bta,
                     float* __restrict__ x_embed, uint16_t* __restrict__ xnorm) {
    __shared__ float red[8];
    int tt = blockIdx.x;
    int c  = threadIdx.x;
    int b  = tt >> 11;
    int l  = tt & (SEQ - 1);
    float v = x[((size_t)b * D_MODEL + c) * SEQ + l];
    x_embed[(size_t)tt * D_MODEL + c] = v;
    float mean = block_sum_256(v, red) * (1.f / D_MODEL);
    float d    = v - mean;
    float var  = block_sum_256(d * d, red) * (1.f / D_MODEL);
    float nv   = d * rsqrtf(var + 1e-5f) * g[c] + bta[c];
    xnorm[(size_t)tt * D_MODEL + c] = f2bf(nv);
}

// ---------------------------------------------------------------------------
// Depthwise causal conv(4) + SiLU.  Reads x_in = xz[:, :512].
// ---------------------------------------------------------------------------
__global__ __launch_bounds__(256)
void conv_silu_kernel(const float* __restrict__ xz,
                      const float* __restrict__ cw, const float* __restrict__ cb,
                      float* __restrict__ xc_f, uint16_t* __restrict__ xc_bf) {
    int idx = blockIdx.x * 256 + threadIdx.x;      // over TOKENS * D_INNER
    int e   = idx & (D_INNER - 1);
    int tt  = idx >> 9;
    int l   = tt & (SEQ - 1);
    float acc = cb[e];
    #pragma unroll
    for (int j = 0; j < D_CONV; ++j) {
        int ls = l - (D_CONV - 1) + j;
        float xv = (ls >= 0) ? xz[(size_t)(tt - (D_CONV - 1) + j) * (2 * D_INNER) + e] : 0.f;
        acc += cw[e * D_CONV + j] * xv;
    }
    float s = acc / (1.f + __expf(-acc));
    xc_f[idx]  = s;
    xc_bf[idx] = f2bf(s);
}

// ---------------------------------------------------------------------------
// delta = softplus(dt @ dt_proj_w^T + dt_proj_b)    (K=16: plain VALU)
// ---------------------------------------------------------------------------
__global__ __launch_bounds__(256)
void dtproj_softplus_kernel(const float* __restrict__ x_dbl,
                            const float* __restrict__ dtw, const float* __restrict__ dtb,
                            float* __restrict__ delta) {
    int idx = blockIdx.x * 256 + threadIdx.x;      // over TOKENS * D_INNER
    int e   = idx & (D_INNER - 1);
    int tt  = idx >> 9;
    const float* dt = x_dbl + (size_t)tt * 48;
    float acc = dtb[e];
    #pragma unroll
    for (int r = 0; r < DT_RANK; ++r) acc += dt[r] * dtw[e * DT_RANK + r];
    delta[idx] = (acc > 20.f) ? acc : log1pf(__expf(acc));
}

// ---------------------------------------------------------------------------
// Selective scan: one lane per (b, channel, state). 16384 lanes = 512 waves.
// 16-lane shfl_xor tree reduces the state dim; lane s==0 applies skip + SiLU
// gate and emits bf16 A-operand for the out_proj WMMA GEMM.
// ---------------------------------------------------------------------------
__global__ __launch_bounds__(256)
void scan_kernel(const float* __restrict__ xc, const float* __restrict__ delta,
                 const float* __restrict__ x_dbl, const float* __restrict__ A_log,
                 const float* __restrict__ D_skip, const float* __restrict__ xz,
                 uint16_t* __restrict__ yg) {
    int gid = blockIdx.x * 256 + threadIdx.x;      // 0 .. B*D_INNER*D_STATE-1
    int s   = gid & (D_STATE - 1);
    int ch  = gid >> 4;                             // 0 .. 1023
    int b   = ch >> 9;
    int e   = ch & (D_INNER - 1);

    float Ae = -__expf(A_log[e * D_STATE + s]);
    float Dv = D_skip[e];
    float h  = 0.f;

    for (int l = 0; l < SEQ; ++l) {
        int tt   = b * SEQ + l;
        float d  = delta[(size_t)tt * D_INNER + e];
        float xv = xc[(size_t)tt * D_INNER + e];
        float Bt = x_dbl[(size_t)tt * 48 + DT_RANK + s];
        float Ct = x_dbl[(size_t)tt * 48 + DT_RANK + D_STATE + s];
        h = __expf(d * Ae) * h + (d * xv) * Bt;
        float y = h * Ct;
        y += __shfl_xor(y, 1, 32);
        y += __shfl_xor(y, 2, 32);
        y += __shfl_xor(y, 4, 32);
        y += __shfl_xor(y, 8, 32);
        if (s == 0) {
            float zv = xz[(size_t)tt * (2 * D_INNER) + D_INNER + e];
            float sg = zv / (1.f + __expf(-zv));
            yg[(size_t)tt * D_INNER + e] = f2bf((y + xv * Dv) * sg);
        }
    }
}

// ---------------------------------------------------------------------------
// Residual add + LayerNorm2 -> bf16 A-operand for MLP.
// ---------------------------------------------------------------------------
__global__ __launch_bounds__(256)
void ln2_kernel(const float* __restrict__ mout, const float* __restrict__ x_embed,
                const float* __restrict__ g, const float* __restrict__ bta,
                uint16_t* __restrict__ x2) {
    __shared__ float red[8];
    int tt = blockIdx.x;
    int c  = threadIdx.x;
    float v = mout[(size_t)tt * D_MODEL + c] + x_embed[(size_t)tt * D_MODEL + c];
    float mean = block_sum_256(v, red) * (1.f / D_MODEL);
    float d    = v - mean;
    float var  = block_sum_256(d * d, red) * (1.f / D_MODEL);
    x2[(size_t)tt * D_MODEL + c] = f2bf(d * rsqrtf(var + 1e-5f) * g[c] + bta[c]);
}

// ---------------------------------------------------------------------------
// GELU(exact) with p1 bias -> bf16 A-operand for p2 GEMM.
// ---------------------------------------------------------------------------
__global__ __launch_bounds__(256)
void gelu_kernel(const float* __restrict__ hpre, const float* __restrict__ p1b,
                 uint16_t* __restrict__ hbf) {
    int idx = blockIdx.x * 256 + threadIdx.x;      // over TOKENS * OUT_DIM
    int o   = idx & (OUT_DIM - 1);
    float v = hpre[idx] + p1b[o];
    float gl = 0.5f * v * (1.f + erff(v * 0.70710678118654752f));
    hbf[idx] = f2bf(gl);
}

// ---------------------------------------------------------------------------
// p2 bias + transpose (T,C) -> (B,C,L) output.
// ---------------------------------------------------------------------------
__global__ __launch_bounds__(256)
void bias_transpose_kernel(const float* __restrict__ opre, const float* __restrict__ p2b,
                           float* __restrict__ out) {
    int idx = blockIdx.x * 256 + threadIdx.x;      // over TOKENS * D_MODEL
    int c   = idx & (D_MODEL - 1);
    int tt  = idx >> 8;
    int b   = tt >> 11;
    int l   = tt & (SEQ - 1);
    out[((size_t)b * D_MODEL + c) * SEQ + l] = opre[idx] + p2b[c];
}

// ---------------------------------------------------------------------------
// Host launcher
// ---------------------------------------------------------------------------
extern "C" void kernel_launch(void* const* d_in, const int* in_sizes, int n_in,
                              void* d_out, int out_size, void* d_ws, size_t ws_size,
                              hipStream_t stream) {
    const float* x        = (const float*)d_in[0];
    const float* ln_g     = (const float*)d_in[1];
    const float* ln_b     = (const float*)d_in[2];
    const float* in_pw    = (const float*)d_in[3];   // (1024,256)
    const float* conv_w   = (const float*)d_in[4];   // (512,4)
    const float* conv_b   = (const float*)d_in[5];
    const float* x_pw     = (const float*)d_in[6];   // (48,512)
    const float* dt_pw    = (const float*)d_in[7];   // (512,16)
    const float* dt_pb    = (const float*)d_in[8];
    const float* A_log    = (const float*)d_in[9];   // (512,16)
    const float* D_skip   = (const float*)d_in[10];
    const float* out_pw   = (const float*)d_in[11];  // (256,512)
    const float* p1_w     = (const float*)d_in[12];  // (1024,256)
    const float* p1_b     = (const float*)d_in[13];
    const float* p2_w     = (const float*)d_in[14];  // (256,1024)
    const float* p2_b     = (const float*)d_in[15];
    float* out            = (float*)d_out;

    // Workspace carve-up (all 256B-aligned; total ~87 MB, L2-resident)
    char*  ws = (char*)d_ws;
    size_t o  = 0;
    auto alloc = [&](size_t bytes) { size_t r = o; o += (bytes + 255) & ~(size_t)255; return r; };

    float*    x_embed = (float*)   (ws + alloc((size_t)TOKENS * D_MODEL * 4));
    uint16_t* xnorm   = (uint16_t*)(ws + alloc((size_t)TOKENS * D_MODEL * 2));
    uint16_t* w_in    = (uint16_t*)(ws + alloc((size_t)2 * D_INNER * D_MODEL * 2));
    uint16_t* w_xp    = (uint16_t*)(ws + alloc((size_t)48 * D_INNER * 2));
    uint16_t* w_out   = (uint16_t*)(ws + alloc((size_t)D_MODEL * D_INNER * 2));
    uint16_t* w_p1    = (uint16_t*)(ws + alloc((size_t)OUT_DIM * D_MODEL * 2));
    uint16_t* w_p2    = (uint16_t*)(ws + alloc((size_t)D_MODEL * OUT_DIM * 2));
    float*    xz      = (float*)   (ws + alloc((size_t)TOKENS * 2 * D_INNER * 4));
    float*    xc_f    = (float*)   (ws + alloc((size_t)TOKENS * D_INNER * 4));
    uint16_t* xc_bf   = (uint16_t*)(ws + alloc((size_t)TOKENS * D_INNER * 2));
    float*    x_dbl   = (float*)   (ws + alloc((size_t)TOKENS * 48 * 4));
    float*    delta   = (float*)   (ws + alloc((size_t)TOKENS * D_INNER * 4));
    uint16_t* yg      = (uint16_t*)(ws + alloc((size_t)TOKENS * D_INNER * 2));
    float*    mout    = (float*)   (ws + alloc((size_t)TOKENS * D_MODEL * 4));
    uint16_t* x2      = (uint16_t*)(ws + alloc((size_t)TOKENS * D_MODEL * 2));
    float*    hpre    = (float*)   (ws + alloc((size_t)TOKENS * OUT_DIM * 4));
    uint16_t* hbf     = (uint16_t*)(ws + alloc((size_t)TOKENS * OUT_DIM * 2));
    float*    opre    = (float*)   (ws + alloc((size_t)TOKENS * D_MODEL * 4));

    // 0) weight conversions f32->bf16
    cvt_bf16_kernel<<<(2 * D_INNER * D_MODEL) / 256, 256, 0, stream>>>(in_pw,  w_in,  2 * D_INNER * D_MODEL);
    cvt_bf16_kernel<<<(48 * D_INNER) / 256,          256, 0, stream>>>(x_pw,   w_xp,  48 * D_INNER);
    cvt_bf16_kernel<<<(D_MODEL * D_INNER) / 256,     256, 0, stream>>>(out_pw, w_out, D_MODEL * D_INNER);
    cvt_bf16_kernel<<<(OUT_DIM * D_MODEL) / 256,     256, 0, stream>>>(p1_w,   w_p1,  OUT_DIM * D_MODEL);
    cvt_bf16_kernel<<<(D_MODEL * OUT_DIM) / 256,     256, 0, stream>>>(p2_w,   w_p2,  D_MODEL * OUT_DIM);

    // 1) embed transpose + LN1
    ln_embed_kernel<<<TOKENS, 256, 0, stream>>>(x, ln_g, ln_b, x_embed, xnorm);

    // Waves per GEMM = (M/16) * (N/16/NT); 8 waves per 256-thread block.

    // 2) in_proj: (4096,256)x(1024,256)^T -> xz (4096,1024)   [WMMA, NT=4]
    wmma_gemm_bf16<4><<<(TOKENS / 16) * ((2 * D_INNER) / 16 / 4) / 8, 256, 0, stream>>>(
        xnorm, w_in, xz, TOKENS, 2 * D_INNER, D_MODEL);

    // 3) depthwise conv(4) + SiLU
    conv_silu_kernel<<<(TOKENS * D_INNER) / 256, 256, 0, stream>>>(xz, conv_w, conv_b, xc_f, xc_bf);

    // 4) x_proj: (4096,512)x(48,512)^T -> x_dbl (4096,48)     [WMMA, NT=3]
    wmma_gemm_bf16<3><<<(TOKENS / 16) * (48 / 16 / 3) / 8, 256, 0, stream>>>(
        xc_bf, w_xp, x_dbl, TOKENS, 48, D_INNER);

    // 5) dt_proj + softplus
    dtproj_softplus_kernel<<<(TOKENS * D_INNER) / 256, 256, 0, stream>>>(x_dbl, dt_pw, dt_pb, delta);

    // 6) selective scan + D-skip + SiLU(z) gate
    scan_kernel<<<(B_SZ * D_INNER * D_STATE) / 256, 256, 0, stream>>>(
        xc_f, delta, x_dbl, A_log, D_skip, xz, yg);

    // 7) out_proj: (4096,512)x(256,512)^T -> mout (4096,256)  [WMMA, NT=4]
    wmma_gemm_bf16<4><<<(TOKENS / 16) * (D_MODEL / 16 / 4) / 8, 256, 0, stream>>>(
        yg, w_out, mout, TOKENS, D_MODEL, D_INNER);

    // 8) residual + LN2
    ln2_kernel<<<TOKENS, 256, 0, stream>>>(mout, x_embed, ln_g, ln_b, x2);

    // 9) p1: (4096,256)x(1024,256)^T -> hpre (4096,1024)      [WMMA, NT=4]
    wmma_gemm_bf16<4><<<(TOKENS / 16) * (OUT_DIM / 16 / 4) / 8, 256, 0, stream>>>(
        x2, w_p1, hpre, TOKENS, OUT_DIM, D_MODEL);

    // 10) bias + exact GELU
    gelu_kernel<<<(TOKENS * OUT_DIM) / 256, 256, 0, stream>>>(hpre, p1_b, hbf);

    // 11) p2: (4096,1024)x(256,1024)^T -> opre (4096,256)     [WMMA, NT=4]
    wmma_gemm_bf16<4><<<(TOKENS / 16) * (D_MODEL / 16 / 4) / 8, 256, 0, stream>>>(
        hbf, w_p2, opre, TOKENS, D_MODEL, OUT_DIM);

    // 12) bias + transpose back to (B,C,L)
    bias_transpose_kernel<<<(TOKENS * D_MODEL) / 256, 256, 0, stream>>>(opre, p2_b, out);

    (void)in_sizes; (void)n_in; (void)out_size; (void)ws_size;
}
